// MolecularGNN_41532333752537
// MI455X (gfx1250) — matrix operand order repaired and checked by reading, not compiled, not run on hardware
//
#include <hip/hip_runtime.h>

#define HID  64
#define OUTF 15

typedef __attribute__((ext_vector_type(16))) _Float16 v16h;
typedef __attribute__((ext_vector_type(8)))  float    v8f;

// ---------------------------------------------------------------------------
// Degree accumulation: deg[row] += 1 per edge (self-loop +1 added in k_dis).
// ---------------------------------------------------------------------------
__global__ void k_degree(const int* __restrict__ rows, float* __restrict__ deg, int nEdges) {
    int e = blockIdx.x * blockDim.x + threadIdx.x;
    if (e < nEdges) atomicAdd(&deg[rows[e]], 1.0f);
}

// dis = rsqrt(deg + 1)  (in place over the deg buffer)
__global__ void k_dis(float* __restrict__ dis, int nNodes) {
    int i = blockIdx.x * blockDim.x + threadIdx.x;
    if (i < nNodes) dis[i] = rsqrtf(dis[i] + 1.0f);
}

// ---------------------------------------------------------------------------
// WMMA GEMM: HW = act(H) @ W   (act = ReLU or identity; H = buffer or emb[x])
// Also fuses the self-loop init:  Aout = HW * dis^2 + bias.
// Block = 256 threads = 8 waves; each wave computes a 16-node x 64-col tile
// with 8x v_wmma_f32_16x16x32_f16 (2 K-steps x 4 N-tiles), f32 accumulate.
//
// W is pre-packed into LDS in the native WMMA B-operand lane layout so each
// tile is a single contiguous 32B v16h load per lane (2x ds_load_b128).
// USE_EMB / RELU are compile-time so the inner load loop has no selects.
//
// Packed index: tile t = kt*4+nt, lane l, half j  ->
//   n = nt*16 + (l&15),  k = kt*32 + (l>>4)*8 + (j&7) + (j>>3)*16
// ---------------------------------------------------------------------------
template <bool USE_EMB, bool RELU>
__global__ void k_gemm(const float* __restrict__ Hin,
                       const float* __restrict__ emb,
                       const int*   __restrict__ xtype,
                       const float* __restrict__ W,
                       const float* __restrict__ bias,
                       const float* __restrict__ dis,
                       float* __restrict__ HW,
                       float* __restrict__ Aout,
                       int nNodes) {
    __shared__ _Float16 Wpk[8 * 32 * 16];       // 8 KB: 8 tiles x 32 lanes x 16 halves
    int tid = threadIdx.x;
    #pragma unroll
    for (int p = tid; p < 8 * 32 * 16; p += 256) {
        int j  = p & 15;
        int l  = (p >> 4) & 31;
        int t  = p >> 9;
        int kt = t >> 2, nt = t & 3;
        int k  = kt * 32 + (l >> 4) * 8 + (j & 7) + (j >> 3) * 16;
        int n  = nt * 16 + (l & 15);
        Wpk[p] = (_Float16)W[k * HID + n];
    }
    __syncthreads();

    int wave = tid >> 5;
    int lane = tid & 31;
    int nodeBase = (blockIdx.x * 8 + wave) * 16;
    if (nodeBase >= nNodes) return;             // wave-uniform: EXEC stays full

    int halfSel = lane >> 4;                    // 0: lanes 0-15, 1: lanes 16-31
    int laneM   = lane & 15;

    // A-operand source row (ISA 7.12.2: lanes 0-15 and 16-31 both hold M = laneM)
    int rowA = nodeBase + laneM;
    bool rowValid = rowA < nNodes;
    const float* srcRow = nullptr;
    if (rowValid)
        srcRow = USE_EMB ? (emb + (size_t)xtype[rowA] * HID)
                         : (Hin + (size_t)rowA * HID);

    v8f acc[4] = {};                            // 4 N-tiles of 16 cols

    #pragma unroll
    for (int kt = 0; kt < 2; ++kt) {            // K = 64 in two 32-wide steps
        int kbase = kt * 32;
        // A 16x32 f16: lanes 0-15 hold K {kb..kb+7, kb+16..kb+23},
        //              lanes 16-31 hold K {kb+8..kb+15, kb+24..kb+31}
        v16h a = {};
        if (rowValid) {
            int k0 = kbase + (halfSel ? 8 : 0);
            #pragma unroll
            for (int i = 0; i < 8; ++i) {
                float v0 = srcRow[k0 + i];
                float v1 = srcRow[k0 + 16 + i];
                if (RELU) { v0 = fmaxf(v0, 0.0f); v1 = fmaxf(v1, 0.0f); }
                a[i]     = (_Float16)v0;
                a[8 + i] = (_Float16)v1;
            }
        }
        #pragma unroll
        for (int nt = 0; nt < 4; ++nt) {
            // One contiguous 32B LDS read: this lane's 16 halves of tile (kt,nt)
            v16h bm = *(const v16h*)&Wpk[(((kt * 4 + nt) * 32) + lane) * 16];
            acc[nt] = __builtin_amdgcn_wmma_f32_16x16x32_f16(
                false, a, false, bm, (short)0, acc[nt], false, false);
        }
    }

    // C/D layout: lane = column (laneM), VGPR r = row r (lanes 0-15) / r+8.
    int rbase = nodeBase + (halfSel ? 8 : 0);
    #pragma unroll
    for (int r = 0; r < 8; ++r) {
        int node = rbase + r;
        if (node < nNodes) {
            float d = dis[node];                // inv_deg = dis^2 (one load per row)
            float idg = d * d;
            #pragma unroll
            for (int nt = 0; nt < 4; ++nt) {
                int col = nt * 16 + laneM;
                float v = acc[nt][r];
                HW[(size_t)node * HID + col]   = v;
                Aout[(size_t)node * HID + col] = v * idg + bias[col];
            }
        }
    }
}

// ---------------------------------------------------------------------------
// Edge aggregation: Aout[row] += HW[col] * dis[row]*dis[col].
// 32 lanes per edge, float2 per lane (coalesced 256B loads); buffers are
// L2-resident (25.6 MB each vs 192 MB L2) so atomics resolve in L2.
// The edge index / degree loads are wave-uniform -> force them onto the
// scalar path via readfirstlane so the vector pipe only carries payload.
// ---------------------------------------------------------------------------
__global__ void k_edge(const int* __restrict__ rows, const int* __restrict__ cols,
                       const float* __restrict__ dis,
                       const float* __restrict__ HW, float* __restrict__ Aout,
                       int nEdges) {
    int e = __builtin_amdgcn_readfirstlane(blockIdx.x * 8 + (threadIdx.x >> 5));
    int lane = threadIdx.x & 31;
    if (e >= nEdges) return;
    int r = __builtin_amdgcn_readfirstlane(rows[e]);
    int c = __builtin_amdgcn_readfirstlane(cols[e]);
    float ne = dis[r] * dis[c];
    float2 v = ((const float2*)(HW + (size_t)c * HID))[lane];
    float* dst = Aout + (size_t)r * HID + 2 * lane;
    atomicAdd(dst,     v.x * ne);
    atomicAdd(dst + 1, v.y * ne);
}

// ---------------------------------------------------------------------------
// Global mean-pool front half: ReLU(h3) segment-summed by graph + counts.
// ---------------------------------------------------------------------------
__global__ void k_pool(const float* __restrict__ H, const int* __restrict__ batch,
                       float* __restrict__ sums, float* __restrict__ cnts,
                       int nNodes) {
    int node = __builtin_amdgcn_readfirstlane(blockIdx.x * 8 + (threadIdx.x >> 5));
    int lane = threadIdx.x & 31;
    if (node >= nNodes) return;
    int g = __builtin_amdgcn_readfirstlane(batch[node]);
    float2 v = ((const float2*)(H + (size_t)node * HID))[lane];
    v.x = fmaxf(v.x, 0.0f);
    v.y = fmaxf(v.y, 0.0f);
    float* dst = sums + (size_t)g * HID + 2 * lane;
    atomicAdd(dst,     v.x);
    atomicAdd(dst + 1, v.y);
    if (lane == 0) atomicAdd(&cnts[g], 1.0f);
}

// ---------------------------------------------------------------------------
// Head: per graph, mean -> concat conc-emb -> ReLU(comb@Wf1+bf1) -> @Wf2+bf2.
// One 64-thread block per graph (tiny: ~9K MACs/graph).
// ---------------------------------------------------------------------------
__global__ void k_head(const float* __restrict__ sums, const float* __restrict__ cnts,
                       const float* __restrict__ conc,
                       const float* __restrict__ Wc,  const float* __restrict__ bc,
                       const float* __restrict__ Wf1, const float* __restrict__ bf1,
                       const float* __restrict__ Wf2, const float* __restrict__ bf2,
                       float* __restrict__ out, int nGraphs) {
    __shared__ float comb[2 * HID];
    __shared__ float hid[HID];
    int g = blockIdx.x;
    if (g >= nGraphs) return;
    int t = threadIdx.x;                        // 0..63
    float cnt = fmaxf(cnts[g], 1.0f);
    comb[t]       = sums[(size_t)g * HID + t] / cnt;
    comb[HID + t] = conc[g] * Wc[t] + bc[t];
    __syncthreads();
    float acc = bf1[t];
    #pragma unroll 8
    for (int i = 0; i < 2 * HID; ++i) acc += comb[i] * Wf1[i * HID + t];
    hid[t] = fmaxf(acc, 0.0f);
    __syncthreads();
    if (t < OUTF) {
        float o = bf2[t];
        #pragma unroll 8
        for (int k = 0; k < HID; ++k) o += hid[k] * Wf2[k * OUTF + t];
        out[(size_t)g * OUTF + t] = o;
    }
}

// ---------------------------------------------------------------------------
extern "C" void kernel_launch(void* const* d_in, const int* in_sizes, int n_in,
                              void* d_out, int out_size, void* d_ws, size_t ws_size,
                              hipStream_t stream) {
    const int*   x     = (const int*)  d_in[0];
    const int*   ei    = (const int*)  d_in[1];
    const int*   batch = (const int*)  d_in[2];
    const float* conc  = (const float*)d_in[3];
    const float* emb   = (const float*)d_in[4];
    const float* W1  = (const float*)d_in[5];  const float* b1  = (const float*)d_in[6];
    const float* W2  = (const float*)d_in[7];  const float* b2  = (const float*)d_in[8];
    const float* W3  = (const float*)d_in[9];  const float* b3  = (const float*)d_in[10];
    const float* Wc  = (const float*)d_in[11]; const float* bc  = (const float*)d_in[12];
    const float* Wf1 = (const float*)d_in[13]; const float* bf1 = (const float*)d_in[14];
    const float* Wf2 = (const float*)d_in[15]; const float* bf2 = (const float*)d_in[16];
    float* out = (float*)d_out;

    const int nNodes  = in_sizes[0];
    const int nEdges  = in_sizes[1] / 2;
    const int nGraphs = in_sizes[3];
    const int* rows = ei;                 // edge_index[0, :]
    const int* cols = ei + nEdges;        // edge_index[1, :]

    // Workspace carve-out (256B aligned): dis + 2 node-feature bufs + pool accum.
    char* ws = (char*)d_ws;
    size_t o = 0;
    auto take = [&](size_t nbytes) -> char* {
        char* p = ws + o;
        o += (nbytes + 255) & ~(size_t)255;
        return p;
    };
    float* dis  = (float*)take((size_t)nNodes * sizeof(float));
    float* bufA = (float*)take((size_t)nNodes * HID * sizeof(float));   // h / agg-out
    float* bufB = (float*)take((size_t)nNodes * HID * sizeof(float));   // hw scratch
    float* sums = (float*)take((size_t)nGraphs * HID * sizeof(float));
    float* cnts = (float*)take((size_t)nGraphs * sizeof(float));
    (void)ws_size; (void)n_in; (void)out_size;

    hipMemsetAsync(dis,  0, (size_t)nNodes * sizeof(float), stream);
    hipMemsetAsync(sums, 0, (size_t)nGraphs * HID * sizeof(float), stream);
    hipMemsetAsync(cnts, 0, (size_t)nGraphs * sizeof(float), stream);

    k_degree<<<(nEdges + 255) / 256, 256, 0, stream>>>(rows, dis, nEdges);
    k_dis   <<<(nNodes + 255) / 256, 256, 0, stream>>>(dis, nNodes);

    const int gemmBlocks = (nNodes + 127) / 128;   // 128 nodes per 8-wave block
    const int edgeBlocks = (nEdges + 7) / 8;       // 8 edges per block
    const int poolBlocks = (nNodes + 7) / 8;

    // Layer 1: h = emb[x];  HW = h@W1;  A = HW*dis^2 + b1;  A += scatter(HW)
    k_gemm<true, false><<<gemmBlocks, 256, 0, stream>>>(bufA, emb, x, W1, b1, dis,
                                                        bufB, bufA, nNodes);
    k_edge<<<edgeBlocks, 256, 0, stream>>>(rows, cols, dis, bufB, bufA, nEdges);

    // Layer 2 (ReLU fused into GEMM input load)
    k_gemm<false, true><<<gemmBlocks, 256, 0, stream>>>(bufA, emb, x, W2, b2, dis,
                                                        bufB, bufA, nNodes);
    k_edge<<<edgeBlocks, 256, 0, stream>>>(rows, cols, dis, bufB, bufA, nEdges);

    // Layer 3
    k_gemm<false, true><<<gemmBlocks, 256, 0, stream>>>(bufA, emb, x, W3, b3, dis,
                                                        bufB, bufA, nNodes);
    k_edge<<<edgeBlocks, 256, 0, stream>>>(rows, cols, dis, bufB, bufA, nEdges);

    // Mean-pool (ReLU fused) + head MLP
    k_pool<<<poolBlocks, 256, 0, stream>>>(bufA, batch, sums, cnts, nNodes);
    k_head<<<nGraphs, 64, 0, stream>>>(sums, cnts, conc, Wc, bc, Wf1, bf1, Wf2, bf2,
                                       out, nGraphs);
}
